// DeepONetGRU_78932908966149
// MI455X (gfx1250) — matrix operand order repaired and checked by reading, not compiled, not run on hardware
//
#include <hip/hip_runtime.h>
#include <math.h>

typedef _Float16 f16;
typedef _Float16 v8h  __attribute__((ext_vector_type(8)));
typedef _Float16 v16h __attribute__((ext_vector_type(16)));
typedef float    v8f  __attribute__((ext_vector_type(8)));

#define B_SZ   1024
#define T_SZ   128
#define NBR    528
#define NBRP   544     // padded to multiple of 32 for K
#define HID    512
#define GH     256
#define G3     768     // 3*GH
#define NST    16

static __device__ __forceinline__ v16h cat16(v8h lo, v8h hi) {
    return __builtin_shufflevector(lo, hi, 0,1,2,3,4,5,6,7,8,9,10,11,12,13,14,15);
}

// A fragment: 16x32 f16 tile from row-major [*, lda] source.
// lane l (0..15): m = m0+l, halves 0..7 = K[k0..k0+7],  halves 8..15 = K[k0+16..k0+23]
// lane l (16..31): m = m0+l-16, K offset +8 in each chunk.  (ISA 7.12.2)
static __device__ __forceinline__ v16h load_afrag(const f16* A, int lda, int m0, int k0, int lane) {
    int m = m0 + (lane & 15);
    int half = lane >> 4;
    const f16* p = A + (size_t)m * lda + k0 + half * 8;
    v8h lo = *(const v8h*)p;
    v8h hi = *(const v8h*)(p + 16);
    return cat16(lo, hi);
}

// B fragment: 32x16 tile, element (k,n) = W[n*ldb + k] (W row-major [N,K]).
// lane l: n = n0 + l%16, halves 0..7 = K[k0+h*8..+7], halves 8..15 = K[k0+16+h*8..+7], h = l>=16.
static __device__ __forceinline__ v16h load_bfrag(const f16* W, int ldb, int n0, int k0, int lane) {
    int n = n0 + (lane & 15);
    int half = lane >> 4;
    const f16* p = W + (size_t)n * ldb + k0 + half * 8;
    v8h lo = *(const v8h*)p;
    v8h hi = *(const v8h*)(p + 16);
    return cat16(lo, hi);
}

static __device__ __forceinline__ v8f wmma_f16(v16h a, v16h b, v8f c) {
    return __builtin_amdgcn_wmma_f32_16x16x32_f16(false, a, false, b, (short)0, c, false, false);
}

static __device__ __forceinline__ float sigmf(float x) { return 1.0f / (1.0f + expf(-x)); }

// ---------------------------------------------------------------------------
// fp32 -> f16 conversion (flat and K-padded variants)
// ---------------------------------------------------------------------------
__global__ void cvt_f16_kernel(const float* __restrict__ src, f16* __restrict__ dst, int n) {
    int i = blockIdx.x * blockDim.x + threadIdx.x;
    if (i < n) dst[i] = (f16)src[i];
}

__global__ void cvt_f16_pad_kernel(const float* __restrict__ src, f16* __restrict__ dst,
                                   int rows, int scols, int dcols) {
    int i = blockIdx.x * blockDim.x + threadIdx.x;
    int n = rows * dcols;
    if (i >= n) return;
    int r = i / dcols, c = i - r * dcols;
    dst[i] = (c < scols) ? (f16)src[(size_t)r * scols + c] : (f16)0.0f;
}

// ---------------------------------------------------------------------------
// Trunk MLP: t = 0..127 rows, 1 -> 512 -> 512 x3, ReLU everywhere (final_act).
// 16 workgroups x 8 rows, plain VALU fp32, LDS ping-pong. Output f16 [128,512].
// ---------------------------------------------------------------------------
__global__ void __launch_bounds__(256)
trunk_kernel(const float* __restrict__ Wf, const float* __restrict__ bf,
             const float* __restrict__ Wr, const float* __restrict__ br,
             f16* __restrict__ outH) {
    __shared__ float hin[8 * HID];
    __shared__ float hout[8 * HID];
    int tid = threadIdx.x;
    int r0 = blockIdx.x * 8;

    for (int o = tid; o < 8 * HID; o += 256) {
        int r = o >> 9, j = o & 511;
        float t = (float)(r0 + r);
        hin[o] = fmaxf(t * Wf[j] + bf[j], 0.0f);
    }
    __syncthreads();

    for (int l = 0; l < 3; l++) {
        const float* W = Wr + (size_t)l * HID * HID;
        const float* b = br + (size_t)l * HID;
        for (int o = tid; o < 8 * HID; o += 256) {
            int r = o >> 9, j = o & 511;
            float s = b[j];
            const float* wr = W + (size_t)j * HID;
            const float* hr = hin + r * HID;
            for (int k = 0; k < HID; k++) s += hr[k] * wr[k];
            hout[o] = fmaxf(s, 0.0f);   // trunk: ReLU on every layer incl. last
        }
        __syncthreads();
        for (int o = tid; o < 8 * HID; o += 256) hin[o] = hout[o];
        __syncthreads();
    }
    for (int o = tid; o < 8 * HID; o += 256) {
        int r = o >> 9, j = o & 511;
        outH[(size_t)(r0 + r) * HID + j] = (f16)hin[o];
    }
}

// ---------------------------------------------------------------------------
// Generic WMMA GEMM:  C[M,N] = act(A[M,K] @ W[N,K]^T + bias)
// A,W f16 row-major (lda = ldb = K). Workgroup tile 128x128, 8 waves,
// each wave owns 2(M) x 4(N) 16x16 accumulator tiles. grid = (M/128, N/128).
// ---------------------------------------------------------------------------
__global__ void __launch_bounds__(256)
gemm_f16_kernel(const f16* __restrict__ A, const f16* __restrict__ W,
                const float* __restrict__ colBias, const float* __restrict__ scalBias,
                float* __restrict__ outF, f16* __restrict__ outH,
                int K, int N, int relu) {
    int lane = threadIdx.x & 31;
    int wave = threadIdx.x >> 5;
    int wm = wave & 3;          // 4 waves in M
    int wn = wave >> 2;         // 2 waves in N
    int rowBase = blockIdx.x * 128 + wm * 32;
    int colBase = blockIdx.y * 128 + wn * 64;

    const v8f vzero = {0.f, 0.f, 0.f, 0.f, 0.f, 0.f, 0.f, 0.f};
    v8f acc[2][4];
    for (int i = 0; i < 2; i++)
        for (int j = 0; j < 4; j++) acc[i][j] = vzero;

    for (int k0 = 0; k0 < K; k0 += 32) {
        v16h a0 = load_afrag(A, K, rowBase,      k0, lane);
        v16h a1 = load_afrag(A, K, rowBase + 16, k0, lane);
        v16h b0 = load_bfrag(W, K, colBase,      k0, lane);
        v16h b1 = load_bfrag(W, K, colBase + 16, k0, lane);
        v16h b2 = load_bfrag(W, K, colBase + 32, k0, lane);
        v16h b3 = load_bfrag(W, K, colBase + 48, k0, lane);
        acc[0][0] = wmma_f16(a0, b0, acc[0][0]);
        acc[0][1] = wmma_f16(a0, b1, acc[0][1]);
        acc[0][2] = wmma_f16(a0, b2, acc[0][2]);
        acc[0][3] = wmma_f16(a0, b3, acc[0][3]);
        acc[1][0] = wmma_f16(a1, b0, acc[1][0]);
        acc[1][1] = wmma_f16(a1, b1, acc[1][1]);
        acc[1][2] = wmma_f16(a1, b2, acc[1][2]);
        acc[1][3] = wmma_f16(a1, b3, acc[1][3]);
    }

    float sb = scalBias ? scalBias[0] : 0.0f;
    int rsel = (lane >> 4) * 8;
    int csel = lane & 15;
    for (int mt = 0; mt < 2; mt++) {
        for (int nt = 0; nt < 4; nt++) {
            int col = colBase + nt * 16 + csel;
            float cb = colBias ? colBias[col] : sb;
            for (int v = 0; v < 8; v++) {
                int row = rowBase + mt * 16 + rsel + v;
                float val = acc[mt][nt][v] + cb;
                if (relu) val = fmaxf(val, 0.0f);
                size_t idx = (size_t)row * N + col;
                if (outF) outF[idx] = val;
                else      outH[idx] = (f16)val;
            }
        }
    }
}

// ---------------------------------------------------------------------------
// Fused 2-layer GRU + projection. One workgroup per 32 batch rows, persistent
// across all 128 timesteps. h0/h1 in LDS (f16 WMMA copy + f32 master),
// gate pre-activation buffers gA/gB [32x768] f32 in LDS. Weights streamed
// from L2 as f16. 288 KB dynamic LDS (CDNA5: 320 KB/WGP).
// ---------------------------------------------------------------------------
static __device__ __forceinline__ void gru_gemm(const f16* Asm /*LDS [32,256]*/,
                                                const f16* __restrict__ W /*[768,256]*/,
                                                float* Gout /*LDS [32,768]*/,
                                                int lane, int wave) {
    const v8f vzero = {0.f, 0.f, 0.f, 0.f, 0.f, 0.f, 0.f, 0.f};
    // 2 M-tiles x 48 N-tiles = 96 tiles over 8 waves
    for (int tile = wave; tile < 96; tile += 8) {
        int mt = tile & 1;
        int nt = tile >> 1;
        int m0 = mt * 16, n0 = nt * 16;
        v8f acc = vzero;
        for (int k0 = 0; k0 < GH; k0 += 32) {
            v16h a = load_afrag(Asm, GH, m0, k0, lane);   // ds_read_b128 x2
            v16h b = load_bfrag(W,   GH, n0, k0, lane);   // global (L2) b128 x2
            acc = wmma_f16(a, b, acc);
        }
        int col  = n0 + (lane & 15);
        int rbase = m0 + ((lane >> 4) << 3);
        for (int v = 0; v < 8; v++) Gout[(rbase + v) * G3 + col] = acc[v];
    }
}

__global__ void __launch_bounds__(256)
gru_kernel(const float* __restrict__ seq,     // [1024,128] f32
           const float* __restrict__ wih0,    // [768] (in=1)
           const float* __restrict__ bih0, const float* __restrict__ bhh0,
           const f16*   __restrict__ whh0,    // [768,256] f16
           const f16*   __restrict__ wih1,    // [768,256] f16
           const float* __restrict__ bih1, const float* __restrict__ bhh1,
           const f16*   __restrict__ whh1,    // [768,256] f16
           const float* __restrict__ projW,   // [16,256] f32
           const float* __restrict__ projB,   // [16]
           float* __restrict__ out) {         // [1024,16]
    extern __shared__ char smem[];
    f16*   h0h = (f16*)(smem);                  // 32*256 f16 = 16 KB
    f16*   h1h = (f16*)(smem + 16 * 1024);      // 16 KB
    float* h0f = (float*)(smem + 32 * 1024);    // 32 KB
    float* h1f = (float*)(smem + 64 * 1024);    // 32 KB
    float* gA  = (float*)(smem + 96 * 1024);    // 32*768 f32 = 96 KB
    float* gB  = (float*)(smem + 192 * 1024);   // 96 KB   (total 288 KB)

    int tid  = threadIdx.x;
    int lane = tid & 31;
    int wave = tid >> 5;
    int b0   = blockIdx.x * 32;

    for (int o = tid; o < 32 * GH; o += 256) {
        h0f[o] = 0.0f; h1f[o] = 0.0f;
        h0h[o] = (f16)0.0f; h1h[o] = (f16)0.0f;
    }
    __syncthreads();

    int j = tid;   // 256 threads <-> 256 hidden columns
    float wr0 = wih0[j], wz0 = wih0[j + GH], wn0 = wih0[j + 2 * GH];
    float bir = bih0[j], biz = bih0[j + GH], bin_ = bih0[j + 2 * GH];
    float bhr = bhh0[j], bhz = bhh0[j + GH], bhn = bhh0[j + 2 * GH];
    float b1ir = bih1[j], b1iz = bih1[j + GH], b1in = bih1[j + 2 * GH];
    float b1hr = bhh1[j], b1hz = bhh1[j + GH], b1hn = bhh1[j + 2 * GH];

    for (int t = 0; t < T_SZ; t++) {
        // --- layer 0: gh0 = h0 @ Whh0^T ---
        gru_gemm(h0h, whh0, gA, lane, wave);
        __syncthreads();
        for (int m = 0; m < 32; m++) {
            float x = seq[(size_t)(b0 + m) * T_SZ + t];
            const float* g = gA + m * G3;
            float r = sigmf(x * wr0 + bir + g[j]          + bhr);
            float z = sigmf(x * wz0 + biz + g[j + GH]     + bhz);
            float n = tanhf(x * wn0 + bin_ + r * (g[j + 2 * GH] + bhn));
            float h = (1.0f - z) * n + z * h0f[m * GH + j];
            h0f[m * GH + j] = h;
            h0h[m * GH + j] = (f16)h;
        }
        __syncthreads();
        // --- layer 1: gi1 = h0_new @ Wih1^T ; gh1 = h1 @ Whh1^T ---
        gru_gemm(h0h, wih1, gA, lane, wave);
        gru_gemm(h1h, whh1, gB, lane, wave);
        __syncthreads();
        for (int m = 0; m < 32; m++) {
            const float* gi = gA + m * G3;
            const float* gh = gB + m * G3;
            float r = sigmf(gi[j]          + b1ir + gh[j]          + b1hr);
            float z = sigmf(gi[j + GH]     + b1iz + gh[j + GH]     + b1hz);
            float n = tanhf(gi[j + 2 * GH] + b1in + r * (gh[j + 2 * GH] + b1hn));
            float h = (1.0f - z) * n + z * h1f[m * GH + j];
            h1f[m * GH + j] = h;
            h1h[m * GH + j] = (f16)h;
        }
        __syncthreads();
    }

    // projection: out = h1 @ projW^T + projB   (32x16 per workgroup, VALU)
    for (int o = tid; o < 32 * NST; o += 256) {
        int m = o >> 4, s = o & 15;
        float acc = projB[s];
        const float* w = projW + s * GH;
        const float* h = h1f + m * GH;
        for (int k = 0; k < GH; k++) acc += h[k] * w[k];
        out[(size_t)(b0 + m) * NST + s] = acc;
    }
}

// ---------------------------------------------------------------------------
// Host-side launcher
// ---------------------------------------------------------------------------
extern "C" void kernel_launch(void* const* d_in, const int* in_sizes, int n_in,
                              void* d_out, int out_size, void* d_ws, size_t ws_size,
                              hipStream_t stream) {
    const float* x    = (const float*)d_in[0];
    const float* bWf  = (const float*)d_in[1];
    const float* bbf  = (const float*)d_in[2];
    const float* bWr  = (const float*)d_in[3];
    const float* bbr  = (const float*)d_in[4];
    const float* tWf  = (const float*)d_in[5];
    const float* tbf  = (const float*)d_in[6];
    const float* tWr  = (const float*)d_in[7];
    const float* tbr  = (const float*)d_in[8];
    const float* donb = (const float*)d_in[9];
    const float* wih0 = (const float*)d_in[10];
    const float* whh0 = (const float*)d_in[11];
    const float* bih0 = (const float*)d_in[12];
    const float* bhh0 = (const float*)d_in[13];
    const float* wih1 = (const float*)d_in[14];
    const float* whh1 = (const float*)d_in[15];
    const float* bih1 = (const float*)d_in[16];
    const float* bhh1 = (const float*)d_in[17];
    const float* pW   = (const float*)d_in[18];
    const float* pB   = (const float*)d_in[19];
    float* out = (float*)d_out;

    char* ws = (char*)d_ws;
    size_t off = 0;
    auto alloc = [&](size_t bytes) -> void* {
        void* p = ws + off;
        off = (off + bytes + 255) & ~(size_t)255;
        return p;
    };
    f16*   xh     = (f16*)alloc((size_t)B_SZ * NBRP * 2);   // padded x
    f16*   wfh    = (f16*)alloc((size_t)HID * NBRP * 2);    // padded branch_Wf
    f16*   wrh    = (f16*)alloc((size_t)3 * HID * HID * 2); // branch_Wr
    f16*   trunkh = (f16*)alloc((size_t)T_SZ * HID * 2);    // trunk output f16
    f16*   whh0h  = (f16*)alloc((size_t)G3 * GH * 2);
    f16*   wih1h  = (f16*)alloc((size_t)G3 * GH * 2);
    f16*   whh1h  = (f16*)alloc((size_t)G3 * GH * 2);
    f16*   hA     = (f16*)alloc((size_t)B_SZ * HID * 2);
    f16*   hB     = (f16*)alloc((size_t)B_SZ * HID * 2);
    float* seqf   = (float*)alloc((size_t)B_SZ * T_SZ * 4);
    (void)ws_size; (void)in_sizes; (void)n_in; (void)out_size;

    // --- weight / input conversions to f16 ---
    {
        int n = B_SZ * NBRP;
        cvt_f16_pad_kernel<<<(n + 255) / 256, 256, 0, stream>>>(x, xh, B_SZ, NBR, NBRP);
        n = HID * NBRP;
        cvt_f16_pad_kernel<<<(n + 255) / 256, 256, 0, stream>>>(bWf, wfh, HID, NBR, NBRP);
        n = 3 * HID * HID;
        cvt_f16_kernel<<<(n + 255) / 256, 256, 0, stream>>>(bWr, wrh, n);
        n = G3 * GH;
        cvt_f16_kernel<<<(n + 255) / 256, 256, 0, stream>>>(whh0, whh0h, n);
        cvt_f16_kernel<<<(n + 255) / 256, 256, 0, stream>>>(wih1, wih1h, n);
        cvt_f16_kernel<<<(n + 255) / 256, 256, 0, stream>>>(whh1, whh1h, n);
    }

    // --- trunk MLP (fp32 VALU, outputs f16 [128,512]) ---
    trunk_kernel<<<16, 256, 0, stream>>>(tWf, tbf, tWr, tbr, trunkh);

    // --- branch MLP: 4 WMMA GEMM layers ---
    dim3 blk(256);
    // L0: [1024,544] x [512,544]^T -> hA, ReLU
    gemm_f16_kernel<<<dim3(B_SZ / 128, HID / 128), blk, 0, stream>>>(
        xh, wfh, bbf, nullptr, nullptr, hA, NBRP, HID, 1);
    // L1
    gemm_f16_kernel<<<dim3(B_SZ / 128, HID / 128), blk, 0, stream>>>(
        hA, wrh, bbr, nullptr, nullptr, hB, HID, HID, 1);
    // L2
    gemm_f16_kernel<<<dim3(B_SZ / 128, HID / 128), blk, 0, stream>>>(
        hB, wrh + (size_t)HID * HID, bbr + HID, nullptr, nullptr, hA, HID, HID, 1);
    // L3 (no ReLU)
    gemm_f16_kernel<<<dim3(B_SZ / 128, HID / 128), blk, 0, stream>>>(
        hA, wrh + (size_t)2 * HID * HID, bbr + 2 * HID, nullptr, nullptr, hB, HID, HID, 0);

    // --- DeepONet contraction: seq[1024,128] = branch @ trunk^T + don_bias ---
    gemm_f16_kernel<<<dim3(B_SZ / 128, T_SZ / 128), blk, 0, stream>>>(
        hB, trunkh, nullptr, donb, seqf, nullptr, HID, T_SZ, 0);

    // --- fused 2-layer GRU recurrence + projection ---
    // hipFuncSetAttribute is an idempotent host-side call (not a stream op, so
    // graph-capture safe); call it unconditionally every invocation so
    // kernel_launch does identical work on every call (no static guards).
    const int gru_lds = 288 * 1024;
    hipFuncSetAttribute(reinterpret_cast<const void*>(gru_kernel),
                        hipFuncAttributeMaxDynamicSharedMemorySize, gru_lds);
    gru_kernel<<<B_SZ / 32, 256, gru_lds, stream>>>(
        seqf, wih0, bih0, bhh0, whh0h, wih1h, bih1, bhh1, whh1h, pW, pB, out);
}
